// CLAM_SB_80307298500994
// MI455X (gfx1250) — compile-verified, matrix operand order checked by reading, not run on hardware
//
#include <hip/hip_runtime.h>

// ---------------------------------------------------------------------------
// CLAM-SB forward for MI455X (gfx1250, wave32, WMMA bf16).
//   N=100000 patches, L=1024, D1=512, D2=256, K=8, NC=2
// GEMM1: LDS-staged bf16 conversion, 8 waves x 4 col-tiles per row-tile.
// Attn:  async global->LDS staging (ASYNCcnt), 8 waves x 2 n-tiles.
// Workspace usage: ~105 MB.
// ---------------------------------------------------------------------------

#define CN 100000
#define CL 1024
#define CD1 512
#define CD2 256
#define CK 8
#define CNC 2

typedef __attribute__((ext_vector_type(16))) __bf16 v16bf;
typedef __attribute__((ext_vector_type(8)))  __bf16 v8bf;
typedef __attribute__((ext_vector_type(8)))  float  v8f;

// ---- float <-> order-preserving uint (for atomic max over floats) ----------
__device__ __forceinline__ unsigned f2ord(float f) {
    unsigned u = __float_as_uint(f);
    return (u & 0x80000000u) ? ~u : (u | 0x80000000u);
}
__device__ __forceinline__ float ord2f(unsigned o) {
    unsigned u = (o & 0x80000000u) ? (o & 0x7fffffffu) : ~o;
    return __uint_as_float(u);
}

// ---------------------------------------------------------------------------
// 0) init: A_raw = battn (atomic accumulation target), M = 0, scalars reset
// ---------------------------------------------------------------------------
__global__ void clam_init(float* __restrict__ A_raw, const float* __restrict__ battn,
                          float* __restrict__ M, unsigned* __restrict__ maxkey,
                          float* __restrict__ sumExp) {
    int i = blockIdx.x * blockDim.x + threadIdx.x;
    if (i < CN)  A_raw[i] = battn[0];
    if (i < CD1) M[i] = 0.0f;
    if (i == 0) { *maxkey = 0u; *sumExp = 0.0f; }
}

// ---------------------------------------------------------------------------
// 1) pack fp32 weight [K x Ncols] into bf16 WMMA B-fragments.
//    Fragment (nt, kt): 32 lanes x 16 bf16, element i of lane l is
//    B[kt*32 + (i<8 ? (l/16)*8+i : 16+(l/16)*8+(i-8)), nt*16 + (l%16)]
//    stored contiguously at ((nt*(K/32)+kt)*32 + l)*16.
// ---------------------------------------------------------------------------
__global__ void clam_pack(const float* __restrict__ W, __bf16* __restrict__ Wp,
                          int K, int Ncols) {
    int fid = blockIdx.x * blockDim.x + threadIdx.x;   // one thread = one lane-fragment
    int ktn = K >> 5;
    int total = (Ncols >> 4) * ktn * 32;
    if (fid >= total) return;
    int lane = fid & 31;
    int frag = fid >> 5;
    int nt = frag / ktn;
    int kt = frag % ktn;
    int ncol  = nt * 16 + (lane & 15);
    int khalf = (lane >> 4) * 8;
    __bf16* dst = Wp + (size_t)fid * 16;
    #pragma unroll
    for (int i = 0; i < 16; ++i) {
        int k = kt * 32 + (i < 8 ? khalf + i : 16 + khalf + (i - 8));
        dst[i] = (__bf16)W[(size_t)k * Ncols + ncol];
    }
}

// ---------------------------------------------------------------------------
// 2) GEMM1: h2 = relu(h @ W1 + b1), bf16 out.
//    One block (8 waves) per 16-row tile; h staged through LDS as bf16 once;
//    each wave owns 4 of the 32 column tiles (4 v8f accumulators).
// ---------------------------------------------------------------------------
__global__ __launch_bounds__(256) void clam_gemm1(
        const float* __restrict__ h, const __bf16* __restrict__ W1p,
        const float* __restrict__ b1, __bf16* __restrict__ h2) {
    __shared__ __align__(16) __bf16 tileA[16 * 128];   // 16 rows x 128 k, row-major, 4 KB

    const int t    = threadIdx.x;
    const int lane = t & 31;
    const int wave = t >> 5;
    const int rt   = blockIdx.x;                       // row tile (0..6249)

    const int m     = lane & 15;
    const int khalf = (lane >> 4) * 8;

    // staging coordinates: thread t loads 8 floats of row (t&15), k-octet (t>>4)
    const int srow = t & 15;
    const int sk8  = t >> 4;                            // 0..15
    const float* srcRow = h + (size_t)(rt * 16 + srow) * CL + sk8 * 8;
    __bf16* dstLds = &tileA[srow * 128 + sk8 * 8];

    v8f acc[4] = {v8f{}, v8f{}, v8f{}, v8f{}};

    for (int kc = 0; kc < CL / 128; ++kc) {
        // ---- cooperative stage: 16x128 fp32 -> bf16 in LDS ----
        float4 f0 = *(const float4*)(srcRow + kc * 128);
        float4 f1 = *(const float4*)(srcRow + kc * 128 + 4);
        v8bf pk;
        pk[0]=(__bf16)f0.x; pk[1]=(__bf16)f0.y; pk[2]=(__bf16)f0.z; pk[3]=(__bf16)f0.w;
        pk[4]=(__bf16)f1.x; pk[5]=(__bf16)f1.y; pk[6]=(__bf16)f1.z; pk[7]=(__bf16)f1.w;
        *(v8bf*)dstLds = pk;                            // ds_store_b128
        __syncthreads();

        #pragma unroll
        for (int kt2 = 0; kt2 < 4; ++kt2) {
            const int kt = kc * 4 + kt2;
            // A fragment straight from row-major LDS (two 16-B runs per lane)
            const __bf16* ap = &tileA[m * 128 + kt2 * 32 + khalf];
            v8bf alo = *(const v8bf*)ap;                // ds_load_b128
            v8bf ahi = *(const v8bf*)(ap + 16);
            v16bf a;
            #pragma unroll
            for (int i = 0; i < 8; ++i) { a[i] = alo[i]; a[i + 8] = ahi[i]; }

            #pragma unroll
            for (int c = 0; c < 4; ++c) {
                const int ct = wave + c * 8;
                const __bf16* bp = W1p + ((size_t)(ct * (CL / 32) + kt) * 32 + lane) * 16;
                v8bf blo = *(const v8bf*)bp;
                v8bf bhi = *(const v8bf*)(bp + 8);
                v16bf b;
                #pragma unroll
                for (int i = 0; i < 8; ++i) { b[i] = blo[i]; b[i + 8] = bhi[i]; }
                acc[c] = __builtin_amdgcn_wmma_f32_16x16x32_bf16(
                             false, a, false, b, (short)0, acc[c], false, false);
            }
        }
        __syncthreads();
    }

    const int rhalf = (lane >> 4) * 8;
    #pragma unroll
    for (int c = 0; c < 4; ++c) {
        const int ct = wave + c * 8;
        const float bias = b1[ct * 16 + m];
        #pragma unroll
        for (int j = 0; j < 8; ++j) {
            float v = acc[c][j] + bias;
            v = v > 0.0f ? v : 0.0f;
            int r = rt * 16 + j + rhalf;               // C/D layout: VGPR j, half-wave M split
            h2[(size_t)r * CD1 + ct * 16 + m] = (__bf16)v;
        }
    }
}

// ---------------------------------------------------------------------------
// 3) attention scores: A_raw[r] += sum_n tanh(h2@Wa+ba)*sigmoid(h2@Wb+bb)*Wattn[n]
//    One block per 16-row tile; the 16x512 bf16 tile of h2 is contiguous, so
//    it is staged into LDS with async global->LDS loads (ASYNCcnt).
//    8 waves x 2 n-tiles, Wa/Wb share each A fragment.
// ---------------------------------------------------------------------------
__global__ __launch_bounds__(256) void clam_attn(
        const __bf16* __restrict__ h2, const __bf16* __restrict__ Wap,
        const __bf16* __restrict__ Wbp, const float* __restrict__ ba,
        const float* __restrict__ bb, const float* __restrict__ Wattn,
        float* __restrict__ A_raw) {
    __shared__ __align__(16) __bf16 tileH[16 * CD1];   // 16 KB, rows contiguous

    const int t    = threadIdx.x;
    const int lane = t & 31;
    const int wave = t >> 5;
    const int rt   = blockIdx.x;                       // row tile (0..6249)

    // ---- async stage: 16 KB contiguous global -> LDS, 4 x b128 per thread ----
    const __bf16* gbase = h2 + (size_t)rt * 16 * CD1;
    #pragma unroll
    for (int r = 0; r < 4; ++r) {
        const int chunk = t + r * 256;                 // 1024 x 16-B chunks
        unsigned ldsAddr = (unsigned)(size_t)(&tileH[chunk * 8]);
        const __bf16* gaddr = gbase + chunk * 8;
        asm volatile("global_load_async_to_lds_b128 %0, %1, off"
                     :: "v"(ldsAddr), "v"(gaddr) : "memory");
    }
    asm volatile("s_wait_asynccnt 0x0" ::: "memory");
    __syncthreads();

    const int m     = lane & 15;
    const int khalf = (lane >> 4) * 8;

    v8f accA[2] = {v8f{}, v8f{}};
    v8f accG[2] = {v8f{}, v8f{}};

    for (int kt = 0; kt < CD1 / 32; ++kt) {
        const __bf16* ap = &tileH[m * CD1 + kt * 32 + khalf];
        v8bf alo = *(const v8bf*)ap;                   // ds_load_b128
        v8bf ahi = *(const v8bf*)(ap + 16);
        v16bf a;
        #pragma unroll
        for (int i = 0; i < 8; ++i) { a[i] = alo[i]; a[i + 8] = ahi[i]; }

        #pragma unroll
        for (int c = 0; c < 2; ++c) {
            const int nt = wave + c * 8;
            const size_t fo = ((size_t)(nt * (CD1 / 32) + kt) * 32 + lane) * 16;
            v8bf wlo = *(const v8bf*)(Wap + fo);
            v8bf whi = *(const v8bf*)(Wap + fo + 8);
            v8bf glo = *(const v8bf*)(Wbp + fo);
            v8bf ghi = *(const v8bf*)(Wbp + fo + 8);
            v16bf bA, bG;
            #pragma unroll
            for (int i = 0; i < 8; ++i) {
                bA[i] = wlo[i]; bA[i + 8] = whi[i];
                bG[i] = glo[i]; bG[i + 8] = ghi[i];
            }
            accA[c] = __builtin_amdgcn_wmma_f32_16x16x32_bf16(
                          false, a, false, bA, (short)0, accA[c], false, false);
            accG[c] = __builtin_amdgcn_wmma_f32_16x16x32_bf16(
                          false, a, false, bG, (short)0, accG[c], false, false);
        }
    }

    const int rhalf = (lane >> 4) * 8;
    #pragma unroll
    for (int c = 0; c < 2; ++c) {
        const int n = (wave + c * 8) * 16 + m;
        const float baV = ba[n], bbV = bb[n], wV = Wattn[n];
        #pragma unroll
        for (int j = 0; j < 8; ++j) {
            float av = tanhf(accA[c][j] + baV);
            float gv = 1.0f / (1.0f + expf(-(accG[c][j] + bbV)));
            float v = av * gv * wV;
            // reduce across the 16 n-columns (lanes within half-wave group)
            v += __shfl_xor(v, 8, 16);
            v += __shfl_xor(v, 4, 16);
            v += __shfl_xor(v, 2, 16);
            v += __shfl_xor(v, 1, 16);
            if ((lane & 15) == 0)
                atomicAdd(&A_raw[rt * 16 + j + rhalf], v);
        }
    }
}

// ---------------------------------------------------------------------------
// 4) softmax pieces: global max (ordered-uint atomic), then exp + sum
// ---------------------------------------------------------------------------
__global__ void clam_maxred(const float* __restrict__ A_raw, unsigned* __restrict__ maxkey) {
    __shared__ unsigned s;
    if (threadIdx.x == 0) s = 0u;
    __syncthreads();
    unsigned local = 0u;
    for (int i = blockIdx.x * blockDim.x + threadIdx.x; i < CN; i += gridDim.x * blockDim.x) {
        unsigned k = f2ord(A_raw[i]);
        local = local > k ? local : k;
    }
    atomicMax(&s, local);
    __syncthreads();
    if (threadIdx.x == 0) atomicMax(maxkey, s);
}

__global__ void clam_expsum(const float* __restrict__ A_raw, const unsigned* __restrict__ maxkey,
                            float* __restrict__ expA, float* __restrict__ sumExp) {
    __shared__ float s;
    if (threadIdx.x == 0) s = 0.0f;
    __syncthreads();
    const float mx = ord2f(*maxkey);
    float local = 0.0f;
    for (int i = blockIdx.x * blockDim.x + threadIdx.x; i < CN; i += gridDim.x * blockDim.x) {
        float e = expf(A_raw[i] - mx);
        expA[i] = e;
        local += e;
    }
    atomicAdd(&s, local);
    __syncthreads();
    if (threadIdx.x == 0) atomicAdd(sumExp, s);
}

// ---------------------------------------------------------------------------
// 5) top-8 / bottom-8 by attention: single WG, 16 rounds of packed argmax
// ---------------------------------------------------------------------------
__global__ __launch_bounds__(1024) void clam_topk(const float* __restrict__ A_raw,
                                                  int* __restrict__ sel) {
    __shared__ unsigned long long sBest;
    __shared__ int sSel[16];
    for (int round = 0; round < 16; ++round) {
        if (threadIdx.x == 0) sBest = 0ull;
        __syncthreads();
        const bool bottom = round >= CK;
        const int base = bottom ? CK : 0;
        const int cnt = round - base;
        unsigned long long best = 0ull;
        for (int i = threadIdx.x; i < CN; i += blockDim.x) {
            bool skip = false;
            for (int s2 = 0; s2 < cnt; ++s2)
                if (sSel[base + s2] == i) skip = true;
            if (skip) continue;
            float v = A_raw[i];
            if (bottom) v = -v;
            unsigned long long p = ((unsigned long long)f2ord(v) << 32) | (unsigned)i;
            best = best > p ? best : p;
        }
        atomicMax(&sBest, best);
        __syncthreads();
        if (threadIdx.x == 0) sSel[round] = (int)(sBest & 0xffffffffull);
        __syncthreads();
    }
    if (threadIdx.x < 16) sel[threadIdx.x] = sSel[threadIdx.x];
}

// ---------------------------------------------------------------------------
// 6) instance classifier loss over 16 selected rows -> out[2]
// ---------------------------------------------------------------------------
__global__ void clam_inst(const __bf16* __restrict__ h2, const int* __restrict__ sel,
                          const float* __restrict__ Winst, const float* __restrict__ binst,
                          float* __restrict__ out) {
    __shared__ float sLog[16][2];
    const int t = threadIdx.x;          // 32 threads: row = t&15, class = t>>4
    const int r = t & 15, c = t >> 4;
    const __bf16* hr = h2 + (size_t)sel[r] * CD1;
    float acc = binst[c];
    for (int d = 0; d < CD1; ++d)
        acc += (float)hr[d] * Winst[d * CNC + c];
    sLog[r][c] = acc;
    __syncthreads();
    if (t == 0) {
        float loss = 0.0f;
        for (int rr = 0; rr < 16; ++rr) {
            float l0 = sLog[rr][0], l1 = sLog[rr][1];
            float mx = fmaxf(l0, l1);
            float lse = mx + logf(expf(l0 - mx) + expf(l1 - mx));
            float lt = (rr < CK) ? l1 : l0;     // targets: top->1, bottom->0
            loss += -(lt - lse);
        }
        out[CNC] = loss / 16.0f;
    }
}

// ---------------------------------------------------------------------------
// 7) attention pooling: M[d] = sum_n expA[n]*h2[n,d]  (scaled by 1/sum later)
// ---------------------------------------------------------------------------
__global__ __launch_bounds__(256) void clam_pool(const __bf16* __restrict__ h2,
                                                 const float* __restrict__ expA,
                                                 float* __restrict__ M) {
    const int d0 = threadIdx.x, d1 = threadIdx.x + 256;
    float a0 = 0.0f, a1 = 0.0f;
    for (int r = blockIdx.x; r < CN; r += gridDim.x) {
        const float a = expA[r];
        const __bf16* hr = h2 + (size_t)r * CD1;
        a0 += a * (float)hr[d0];
        a1 += a * (float)hr[d1];
    }
    atomicAdd(&M[d0], a0);
    atomicAdd(&M[d1], a1);
}

// ---------------------------------------------------------------------------
// 8) bag logits: out[0..1] = (M/sumExp) @ Wcls + bcls
// ---------------------------------------------------------------------------
__global__ void clam_final(const float* __restrict__ M, const float* __restrict__ sumExp,
                           const float* __restrict__ Wcls, const float* __restrict__ bcls,
                           float* __restrict__ out) {
    const int lane = threadIdx.x;
    const float inv = 1.0f / *sumExp;
    float a0 = 0.0f, a1 = 0.0f;
    for (int d = lane; d < CD1; d += 32) {
        float m = M[d] * inv;
        a0 += m * Wcls[d * CNC + 0];
        a1 += m * Wcls[d * CNC + 1];
    }
    #pragma unroll
    for (int o = 16; o > 0; o >>= 1) {
        a0 += __shfl_down(a0, o);
        a1 += __shfl_down(a1, o);
    }
    if (lane == 0) { out[0] = a0 + bcls[0]; out[1] = a1 + bcls[1]; }
}

// ---------------------------------------------------------------------------
extern "C" void kernel_launch(void* const* d_in, const int* in_sizes, int n_in,
                              void* d_out, int out_size, void* d_ws, size_t ws_size,
                              hipStream_t stream) {
    const float* h     = (const float*)d_in[0];
    const float* W1    = (const float*)d_in[1];
    const float* b1    = (const float*)d_in[2];
    const float* Wa    = (const float*)d_in[3];
    const float* ba    = (const float*)d_in[4];
    const float* Wb    = (const float*)d_in[5];
    const float* bb    = (const float*)d_in[6];
    const float* Wattn = (const float*)d_in[7];
    const float* battn = (const float*)d_in[8];
    const float* Wcls  = (const float*)d_in[9];
    const float* bcls  = (const float*)d_in[10];
    const float* Winst = (const float*)d_in[11];
    const float* binst = (const float*)d_in[12];
    float* out = (float*)d_out;

    // workspace bump allocator (256-B aligned)
    char* ws = (char*)d_ws;
    size_t off = 0;
    auto bump = [&](size_t bytes) -> char* {
        char* p = ws + off;
        off += (bytes + 255) & ~(size_t)255;
        return p;
    };
    __bf16*  h2     = (__bf16*)bump((size_t)CN * CD1 * 2);       // 102.4 MB
    __bf16*  W1p    = (__bf16*)bump((size_t)CL * CD1 * 2);       // 1 MB
    __bf16*  Wap    = (__bf16*)bump((size_t)CD1 * CD2 * 2);      // 256 KB
    __bf16*  Wbp    = (__bf16*)bump((size_t)CD1 * CD2 * 2);      // 256 KB
    float*   A_raw  = (float*)bump((size_t)CN * 4);
    float*   expA   = (float*)bump((size_t)CN * 4);
    float*   M      = (float*)bump((size_t)CD1 * 4);
    unsigned* maxkey = (unsigned*)bump(64);
    float*   sumExp = (float*)bump(64);
    int*     sel    = (int*)bump(64);

    // 0) init accumulation targets
    clam_init<<<(CN + 255) / 256, 256, 0, stream>>>(A_raw, battn, M, maxkey, sumExp);

    // 1) pack weights to bf16 WMMA fragments
    clam_pack<<<(32 * 32 * 32 + 255) / 256, 256, 0, stream>>>(W1, W1p, CL, CD1);
    clam_pack<<<(16 * 16 * 32 + 255) / 256, 256, 0, stream>>>(Wa, Wap, CD1, CD2);
    clam_pack<<<(16 * 16 * 32 + 255) / 256, 256, 0, stream>>>(Wb, Wbp, CD1, CD2);

    // 2) h2 = relu(h @ W1 + b1)        one block per 16-row tile
    clam_gemm1<<<CN / 16, 256, 0, stream>>>(h, W1p, b1, h2);

    // 3) gated attention scores -> A_raw   one block per 16-row tile
    clam_attn<<<CN / 16, 256, 0, stream>>>(h2, Wap, Wbp, ba, bb, Wattn, A_raw);

    // 4) softmax pieces
    clam_maxred<<<256, 256, 0, stream>>>(A_raw, maxkey);
    clam_expsum<<<256, 256, 0, stream>>>(A_raw, maxkey, expA, sumExp);

    // 5) top/bottom-8 indices
    clam_topk<<<1, 1024, 0, stream>>>(A_raw, sel);

    // 6) instance loss -> out[2]
    clam_inst<<<1, 32, 0, stream>>>(h2, sel, Winst, binst, out);

    // 7) attention pooling
    clam_pool<<<512, 256, 0, stream>>>(h2, expA, M);

    // 8) bag logits -> out[0..1]
    clam_final<<<1, 32, 0, stream>>>(M, sumExp, Wcls, bcls, out);
}
